// HyperPriorDistributedAutoEncoder_53068615910065
// MI455X (gfx1250) — matrix operand order repaired and checked by reading, not compiled, not run on hardware
//
#include <hip/hip_runtime.h>
#include <math.h>

// =====================================================================
// MI455X (gfx1250) HyperPrior distributed AE forward.
// All convolutions (normal + lhs-dilated "transposed") via implicit-GEMM
// v_wmma_f32_16x16x32_f16 (wave32). Weights are repacked once per layer
// into ISA A-fragment order (f16), so the hot loop is:
//   2x global_load_b128 (A frag) + LDS im2col fill + 4x ds_load_b128
//   (two B frags) + 2x v_wmma per wave per 32-K chunk.
// All inner-loop integer div/mod use compile-time divisors (templated).
// =====================================================================

typedef __attribute__((ext_vector_type(16))) _Float16 v16h;
typedef __attribute__((ext_vector_type(8)))  float    v8f;

#define C_CH 192

// -------------------------------------------------------------------
// Weight repack: fp32 OIHW -> f16 fragment-ready layout.
// pack[((t*NC + c)*32 + lane)*16 + j] = W[t*16 + (lane&15)]
//                                        [c*32 + koff(j, lane>>4)]
// koff(j,hi) = (j&7) + 8*((j>>3)*2 + hi)   (ISA 16x32 f16 A layout)
// Rows padded to 64 (grid.y granularity), K padded to 32, zeros filled.
// -------------------------------------------------------------------
__global__ void k_pack_w(const float* __restrict__ w, _Float16* __restrict__ pk,
                         int Cout, int Ktot, int NC, int total) {
  int i = blockIdx.x * blockDim.x + threadIdx.x;
  if (i >= total) return;
  int j    = i & 15;
  int lane = (i >> 4) & 31;
  int rest = i >> 9;                 // t*NC + c
  int cch  = rest % NC;
  int t    = rest / NC;
  int hi = lane >> 4, lc = lane & 15;
  int row  = t * 16 + lc;
  int koff = (j & 7) + 8 * (((j >> 3) << 1) + hi);
  int k    = cch * 32 + koff;
  float v = (row < Cout && k < Ktot) ? w[(size_t)row * Ktot + k] : 0.0f;
  pk[i] = (_Float16)v;
}

// -------------------------------------------------------------------
// Implicit-GEMM conv. Block = 128 thr = 4 waves. N tile = 32 pixels
// (two 16-col sub-tiles), M tile = 64 rows (16 per wave), K chunk = 32.
// Dilated-input coord: r = oh*STRIDE + kh - pad; LD=2 requires r even.
// -------------------------------------------------------------------
template<int KK, int STRIDE, int LD>
__global__ __launch_bounds__(128)
void k_conv_wmma(const float* __restrict__ in, const _Float16* __restrict__ packA,
                 const float* __restrict__ bias, float* __restrict__ out,
                 int B, int Cin, int H, int W, int Cout, int OH, int OW,
                 int padh, int padw, int fuse, int NC, int Ktot)
{
  constexpr int KHW = KK * KK;
  __shared__ _Float16 lds_b[1024];       // two 32x16 fragment-ready subtiles
  const int tid  = threadIdx.x;
  const int wave = tid >> 5;
  const int lane = tid & 31;
  const int hi   = lane >> 4;
  const int lc   = lane & 15;
  const int OHOW = OH * OW;
  const int NP   = B * OHOW;
  const int n0   = blockIdx.x * 32;
  const int tM   = blockIdx.y * 4 + wave;          // 16-row M tile
  const int m0   = tM * 16;
  const int HW   = H * W;

  // ---- per-thread fill-column precompute (fixed for all K chunks) ----
  const int  nn   = tid & 31;
  const int  np   = n0 + nn;
  const bool npok = (np < NP);
  int b = 0, oh = 0, ow = 0;
  if (npok) {                                        // one-time runtime divs
    b = np / OHOW; int prem = np - b * OHOW;
    oh = prem / OW; ow = prem - oh * OW;
  }
  const int    rbase  = oh * STRIDE - padh;
  const int    cbase  = ow * STRIDE - padw;
  const size_t inbase = (size_t)b * Cin * HW;
  const int    lds_wb = (nn >> 4) * 512 + (nn & 15) * 16;

  v8f acc0 = {}, acc1 = {};
  const _Float16* apt = packA + (size_t)tM * NC * 512 + lane * 16;

  for (int cidx = 0; cidx < NC; ++cidx) {
    const int k0 = cidx * 32;
    // ---- cooperative im2col fill (8 k-rows of this thread's pixel) ----
    #pragma unroll
    for (int i = 0; i < 8; ++i) {
      int kk = wave + 4 * i;                        // 0..31, fixed pattern
      int k  = k0 + kk;
      float v = 0.0f;
      if (npok && k < Ktot) {
        int ci  = k / KHW;                          // compile-time divisor
        int rem = k - ci * KHW;
        int kh  = rem / KK;                         // compile-time divisor
        int kw  = rem - kh * KK;
        int r = rbase + kh;
        int c = cbase + kw;
        int ih, iw; bool ok;
        if (LD == 2) {
          ok = (r >= 0) & (c >= 0) & ((r & 1) == 0) & ((c & 1) == 0);
          ih = r >> 1; iw = c >> 1;
        } else { ok = true; ih = r; iw = c; }
        ok = ok & (ih >= 0) & (ih < H) & (iw >= 0) & (iw < W);
        if (ok) v = in[inbase + (size_t)ci * HW + ih * W + iw];
      }
      lds_b[lds_wb + ((kk >> 3) & 1) * 256 + (kk >> 4) * 8 + (kk & 7)] = (_Float16)v;
    }
    __syncthreads();

    v16h a = *(const v16h*)(apt + (size_t)cidx * 512);   // 2x global_load_b128
    const v16h* bp = (const v16h*)lds_b;
    v16h b0 = bp[lane];                                  // 2x ds_load_b128
    v16h b1 = bp[32 + lane];
    if (cidx + 1 < NC)
      __builtin_prefetch(apt + (size_t)(cidx + 1) * 512, 0, 1); // global_prefetch_b8

    acc0 = __builtin_amdgcn_wmma_f32_16x16x32_f16(false, a, false, b0,
                                                  (short)0, acc0, false, false);
    acc1 = __builtin_amdgcn_wmma_f32_16x16x32_f16(false, a, false, b1,
                                                  (short)0, acc1, false, false);
    __syncthreads();
  }

  // ---- write D (ISA f32 C/D layout: M = rr + 8*hi, N = lc) ----
  #pragma unroll
  for (int rr = 0; rr < 8; ++rr) {
    int row = m0 + rr + 8 * hi;
    if (row < Cout) {
      float badd = bias ? bias[row] : 0.0f;
      #pragma unroll
      for (int s = 0; s < 2; ++s) {
        int npo = n0 + s * 16 + lc;
        if (npo < NP) {
          int bb   = npo / OHOW;
          int prem = npo - bb * OHOW;
          float v = (s == 0 ? acc0[rr] : acc1[rr]) + badd;
          if (fuse == 1) v = fmaxf(v, 0.0f);
          out[((size_t)bb * Cout + row) * OHOW + prem] = v;
        }
      }
    }
  }
}

// -------------------------------------------------------------------
// Fused GDN / iGDN: n = sqrt(g @ (x*x) + be); out = x/n (x*n if inverse).
// One block per pixel, 192 threads = channels, x^2 in LDS, float4 g rows.
// -------------------------------------------------------------------
__global__ __launch_bounds__(C_CH)
void k_gdn(const float* __restrict__ x, const float* __restrict__ g,
           const float* __restrict__ be, float* __restrict__ out,
           int HW, int inverse)
{
  __shared__ float sq[C_CH];
  int p = blockIdx.x;
  int c = threadIdx.x;
  int b = p / HW;
  int off = p - b * HW;
  size_t idx = ((size_t)(b * C_CH + c)) * HW + off;
  float v = x[idx];
  sq[c] = v * v;
  __syncthreads();
  float a = be[c];
  const float4* g4 = (const float4*)(g + c * C_CH);
  const float4* s4 = (const float4*)sq;
  #pragma unroll 4
  for (int j = 0; j < C_CH / 4; ++j) {
    float4 gg = g4[j]; float4 ss = s4[j];
    a += gg.x * ss.x + gg.y * ss.y + gg.z * ss.z + gg.w * ss.w;
  }
  float n = sqrtf(a);
  out[idx] = inverse ? v * n : v / n;
}

// ------------------------- elementwise kernels ----------------------
__global__ void k_abs(const float* __restrict__ in, float* __restrict__ out, int n) {
  int i = blockIdx.x * blockDim.x + threadIdx.x;
  if (i < n) out[i] = fabsf(in[i]);
}
__global__ void k_maxc(const float* __restrict__ in, float* __restrict__ out, float bnd, int n) {
  int i = blockIdx.x * blockDim.x + threadIdx.x;
  if (i < n) out[i] = fmaxf(in[i], bnd);
}
__global__ void k_gauss_lik(const float* __restrict__ x, const float* __restrict__ sig,
                            float* __restrict__ out, int n) {
  int i = blockIdx.x * blockDim.x + threadIdx.x;
  if (i < n) {
    const float is2 = 0.70710678118654752440f;
    float s = sig[i];
    float a = (x[i] + 0.5f) / s;
    float b = (x[i] - 0.5f) / s;
    out[i] = 0.5f * (erff(a * is2) - erff(b * is2));
  }
}
__global__ void k_fact_lik(const float* __restrict__ x, const float* __restrict__ s,
                           float* __restrict__ out, int HW, int n) {
  int i = blockIdx.x * blockDim.x + threadIdx.x;
  if (i < n) {
    int c = (i / HW) % C_CH;
    float sc = s[c];
    float a = (x[i] + 0.5f) / sc;
    float b = (x[i] - 0.5f) / sc;
    out[i] = 1.0f / (1.0f + expf(-a)) - 1.0f / (1.0f + expf(-b));
  }
}
__global__ void k_concat(const float* __restrict__ a, const float* __restrict__ b,
                         float* __restrict__ out, int HW, int n) {
  int i = blockIdx.x * blockDim.x + threadIdx.x;
  if (i < n) {
    int per = 2 * C_CH * HW;
    int bb  = i / per;
    int rem = i - bb * per;
    int ch  = rem / HW;
    int off = rem - ch * HW;
    out[i] = (ch < C_CH) ? a[(bb * C_CH + ch) * HW + off]
                         : b[(bb * C_CH + (ch - C_CH)) * HW + off];
  }
}
__global__ void k_post(const float* __restrict__ t, float* __restrict__ out,
                       int H, int W, int n) {
  int i = blockIdx.x * blockDim.x + threadIdx.x;
  if (i < n) {
    int b = i / (H * W);
    int rem = i - b * (H * W);
    int h = rem / W;
    int w = rem - h * W;
    int HW = H * W;
    int hc = h & ~1, wc = w & ~1;
    float Y  = t[(b * 3 + 0) * HW + h * W + w];
    float Cb = t[(b * 3 + 1) * HW + hc * W + wc] - 128.0f;
    float Cr = t[(b * 3 + 2) * HW + hc * W + wc] - 128.0f;
    float R  = Y + 1.402f * Cr;
    float G  = Y - 0.344136f * Cb - 0.714136f * Cr;
    float Bv = Y + 1.772f * Cb;
    const float inv255 = 1.0f / 255.0f;
    out[(b * 3 + 0) * HW + h * W + w] = fminf(fmaxf(R,  0.0f), 255.0f) * inv255;
    out[(b * 3 + 1) * HW + h * W + w] = fminf(fmaxf(G,  0.0f), 255.0f) * inv255;
    out[(b * 3 + 2) * HW + h * W + w] = fminf(fmaxf(Bv, 0.0f), 255.0f) * inv255;
  }
}

// ============================ host side =============================
struct Ctx {
  hipStream_t st;
  const float* const* P;
  char* ws;
  size_t off;
};
static void* c_alloc(Ctx& c, size_t bytes) {
  void* p = c.ws + c.off;
  c.off += (bytes + 255) & ~(size_t)255;
  return p;
}
static float* allocf(Ctx& c, size_t nf) { return (float*)c_alloc(c, nf * 4); }

static inline int out_dim(int H, int K, int s, int plo, int phi, int ld) {
  int D = (H - 1) * ld + 1;
  return (D + plo + phi - K) / s + 1;
}

#define EW_GRID(n) dim3(((n) + 255) / 256), dim3(256)

static void conv(Ctx& c, const float* in, int widx, int bidx, float* out,
                 int B, int Cin, int H, int W, int Cout, int K,
                 int stride, int plo, int phi, int ld, int fuse) {
  const int Ktot = Cin * K * K;
  const int NC   = (Ktot + 31) / 32;
  const int padM = ((Cout + 63) / 64) * 64;
  const size_t npack = (size_t)(padM / 16) * NC * 512;
  _Float16* pk = (_Float16*)c_alloc(c, npack * sizeof(_Float16));
  k_pack_w<<<EW_GRID((int)npack), 0, c.st>>>(c.P[widx], pk, Cout, Ktot, NC, (int)npack);

  int OH = out_dim(H, K, stride, plo, phi, ld);
  int OW = out_dim(W, K, stride, plo, phi, ld);
  int NP = B * OH * OW;
  dim3 grid((NP + 31) / 32, (Cout + 63) / 64);
  const float* bias = c.P[bidx];
  if (K == 5 && stride == 2 && ld == 1)
    k_conv_wmma<5, 2, 1><<<grid, 128, 0, c.st>>>(in, pk, bias, out, B, Cin, H, W,
                                                 Cout, OH, OW, plo, plo, fuse, NC, Ktot);
  else if (K == 3)
    k_conv_wmma<3, 1, 1><<<grid, 128, 0, c.st>>>(in, pk, bias, out, B, Cin, H, W,
                                                 Cout, OH, OW, plo, plo, fuse, NC, Ktot);
  else  // K==5, stride 1, lhs_dilation 2 (transposed conv)
    k_conv_wmma<5, 1, 2><<<grid, 128, 0, c.st>>>(in, pk, bias, out, B, Cin, H, W,
                                                 Cout, OH, OW, plo, plo, fuse, NC, Ktot);
}

static void gdn(Ctx& c, const float* x, int gidx, int beidx, float* out, int HW, int inv) {
  k_gdn<<<4 * HW, C_CH, 0, c.st>>>(x, c.P[gidx], c.P[beidx], out, HW, inv);
}

// parameter index bases (flattened dict-insertion order of setup_inputs)
#define IN_X 0
#define IN_Y 1
#define P_ENC_X 2
#define P_ENC_Y 16
#define P_ENC_W 30
#define P_DEC_X 44
#define P_DEC_Y 58
#define P_HA_X 72
#define P_HS_X 78
#define P_HA_Y 84
#define P_HS_Y 90
#define P_SZX 96
#define P_SZY 97
#define P_SW 98
// enc/dec group: w1 b1 w2 b2 w3 b3 w4 b4 g1 be1 g2 be2 g3 be3
// ha/hs group:   w1 b1 w2 b2 w3 b3

static void encode_net(Ctx& c, const float* img, int base, float* P0, float* P1, float* hout) {
  conv(c, img, base + 0, base + 1, P0, 4,   3, 256, 256, 192, 5, 2, 2, 2, 1, 0);
  gdn(c, P0, base + 8,  base + 9,  P1, 128 * 128, 0);
  conv(c, P1, base + 2, base + 3, P0, 4, 192, 128, 128, 192, 5, 2, 2, 2, 1, 0);
  gdn(c, P0, base + 10, base + 11, P1, 64 * 64, 0);
  conv(c, P1, base + 4, base + 5, P0, 4, 192,  64,  64, 192, 5, 2, 2, 2, 1, 0);
  gdn(c, P0, base + 12, base + 13, P1, 32 * 32, 0);
  conv(c, P1, base + 6, base + 7, hout, 4, 192, 32, 32, 192, 5, 2, 2, 2, 1, 0);
}

static void decode_net(Ctx& c, const float* h, const float* wl, int base,
                       float* catb, float* P0, float* P1) {
  int ncat = 4 * 2 * C_CH * 16 * 16;
  k_concat<<<EW_GRID(ncat), 0, c.st>>>(h, wl, catb, 16 * 16, ncat);
  conv(c, catb, base + 0, base + 1, P0, 4, 384, 16, 16, 192, 5, 1, 2, 3, 2, 0);
  gdn(c, P0, base + 8,  base + 9,  P1, 32 * 32, 1);
  conv(c, P1,   base + 2, base + 3, P0, 4, 192, 32, 32, 192, 5, 1, 2, 3, 2, 0);
  gdn(c, P0, base + 10, base + 11, P1, 64 * 64, 1);
  conv(c, P1,   base + 4, base + 5, P0, 4, 192, 64, 64, 192, 5, 1, 2, 3, 2, 0);
  gdn(c, P0, base + 12, base + 13, P1, 128 * 128, 1);
  conv(c, P1,   base + 6, base + 7, P0, 4, 192, 128, 128, 3, 5, 1, 2, 3, 2, 0);
  // decoded image (4,3,256,256) left in P0
}

static void ha_net(Ctx& c, const float* h, int base, float* S0, float* S1, float* zout) {
  int n16 = 4 * C_CH * 16 * 16;
  k_abs<<<EW_GRID(n16), 0, c.st>>>(h, S0, n16);
  conv(c, S0, base + 0, base + 1, S1, 4, 192, 16, 16, 192, 3, 1, 1, 1, 1, 1);
  conv(c, S1, base + 2, base + 3, S0, 4, 192, 16, 16, 192, 5, 2, 2, 2, 1, 1);
  conv(c, S0, base + 4, base + 5, zout, 4, 192, 8, 8, 192, 5, 2, 2, 2, 1, 0);
}

static void hs_net(Ctx& c, const float* z, int base, float* S0, float* S1, float* sigout) {
  conv(c, z,  base + 0, base + 1, S0, 4, 192, 4, 4, 192, 5, 1, 2, 3, 2, 1);
  conv(c, S0, base + 2, base + 3, S1, 4, 192, 8, 8, 192, 5, 1, 2, 3, 2, 1);
  conv(c, S1, base + 4, base + 5, S0, 4, 192, 16, 16, 192, 3, 1, 1, 1, 1, 0);
  int n16 = 4 * C_CH * 16 * 16;
  k_maxc<<<EW_GRID(n16), 0, c.st>>>(S0, sigout, 0.11f, n16);
}

extern "C" void kernel_launch(void* const* d_in, const int* in_sizes, int n_in,
                              void* d_out, int out_size, void* d_ws, size_t ws_size,
                              hipStream_t stream) {
  (void)in_sizes; (void)n_in; (void)out_size; (void)ws_size;
  Ctx c{stream, (const float* const*)d_in, (char*)d_ws, 0};
  const float* x = c.P[IN_X];
  const float* y = c.P[IN_Y];
  float* out = (float*)d_out;

  float* P0   = allocf(c, 4ull * 192 * 128 * 128);   // 50.3 MB ping
  float* P1   = allocf(c, 4ull * 192 * 128 * 128);   // 50.3 MB pong
  float* S0   = allocf(c, 4ull * 192 * 16 * 16);
  float* S1   = allocf(c, 4ull * 192 * 16 * 16);
  float* hx   = allocf(c, 4ull * 192 * 16 * 16);
  float* hy   = allocf(c, 4ull * 192 * 16 * 16);
  float* wl   = allocf(c, 4ull * 192 * 16 * 16);
  float* sig  = allocf(c, 4ull * 192 * 16 * 16);
  float* sigc = allocf(c, 4ull * 192 * 16 * 16);
  float* zb   = allocf(c, 4ull * 192 * 4 * 4);
  float* zcb  = allocf(c, 4ull * 192 * 4 * 4);
  float* catb = allocf(c, 4ull * 384 * 16 * 16);
  // (packed-weight buffers are bump-allocated per conv call, ~49 MB total)

  const int n_img = 4 * 3 * 256 * 256;
  const int n_16  = 4 * 192 * 16 * 16;
  const int n_4   = 4 * 192 * 4 * 4;
  float* o_px   = out;
  float* o_py   = out + n_img;
  float* o_xlik = out + 2 * n_img;
  float* o_ylik = o_xlik + n_16;
  float* o_zlik = o_ylik + n_16;
  float* o_zlc  = o_zlik + n_4;
  float* o_wlik = o_zlc + n_4;

  encode_net(c, x, P_ENC_X, P0, P1, hx);
  encode_net(c, y, P_ENC_Y, P0, P1, hy);
  encode_net(c, y, P_ENC_W, P0, P1, wl);

  ha_net(c, hx, P_HA_X, S0, S1, zb);
  k_fact_lik<<<EW_GRID(n_4), 0, stream>>>(zb, c.P[P_SZX], o_zlik, 16, n_4);
  hs_net(c, zb, P_HS_X, S0, S1, sig);

  ha_net(c, hy, P_HA_Y, S0, S1, zcb);
  k_fact_lik<<<EW_GRID(n_4), 0, stream>>>(zcb, c.P[P_SZY], o_zlc, 16, n_4);
  hs_net(c, zcb, P_HS_Y, S0, S1, sigc);

  k_gauss_lik<<<EW_GRID(n_16), 0, stream>>>(hx, sig, o_xlik, n_16);
  k_gauss_lik<<<EW_GRID(n_16), 0, stream>>>(hy, sigc, o_ylik, n_16);
  k_fact_lik<<<EW_GRID(n_16), 0, stream>>>(wl, c.P[P_SW], o_wlik, 256, n_16);

  const int n_pix = 4 * 256 * 256;
  decode_net(c, hx, wl, P_DEC_X, catb, P0, P1);
  k_post<<<EW_GRID(n_pix), 0, stream>>>(P0, o_px, 256, 256, n_pix);
  decode_net(c, hy, wl, P_DEC_Y, catb, P0, P1);
  k_post<<<EW_GRID(n_pix), 0, stream>>>(P0, o_py, 256, 256, n_pix);
}
